// TriangleBlock_12738873000127
// MI455X (gfx1250) — matrix operand order repaired and checked by reading, not compiled, not run on hardware
//
#include <hip/hip_runtime.h>
#include <hip/hip_bf16.h>

#define NN 320
#define CZ 128
#define MM (NN * NN)   /* 102400 pair rows */
#define HH 4
#define CA 32

typedef __bf16 bf16;
typedef __attribute__((ext_vector_type(16))) __bf16 v16bf;
typedef __attribute__((ext_vector_type(8)))  __bf16 v8bf;
typedef __attribute__((ext_vector_type(8)))  float  v8f;

__device__ __forceinline__ float sigf(float x) { return 1.0f / (1.0f + __expf(-x)); }

__device__ __forceinline__ v16bf combine16(v8bf lo, v8bf hi) {
  v16bf r;
#pragma unroll
  for (int t = 0; t < 8; t++) { r[t] = lo[t]; r[t + 8] = hi[t]; }
  return r;
}

// A operand (16x32 bf16), row-major source with leading dim ld.
// Per ISA: lanes 0-15 hold row M=lane, K chunks {k0..k0+7, k0+16..k0+23};
// lanes 16-31 hold the {+8, +24} chunks. Chunks are 16B-aligned contiguous.
__device__ __forceinline__ v16bf load_a_frag(const bf16* base, int row0, int ld, int k0) {
  int lane = threadIdx.x & 31;
  int r  = row0 + (lane & 15);
  int g0 = (lane & 16) ? 8 : 0;
  const v8bf* p = reinterpret_cast<const v8bf*>(base + (size_t)r * ld + k0 + g0);
  return combine16(p[0], p[2]);   // p[2] == +16 elements
}

// B operand (32x16 bf16) where Bop[k][n] = src[n0+n][k]: lane holds column n=lane%16,
// K chunks {k0..k0+7, k0+16..k0+23} (+8 for lanes 16-31), matching the A frag.
__device__ __forceinline__ v16bf load_b_frag(const bf16* base, int n0, int ld, int k0) {
  int lane = threadIdx.x & 31;
  int n  = n0 + (lane & 15);
  int kk = k0 + ((lane & 16) ? 8 : 0);
  const v8bf* p = reinterpret_cast<const v8bf*>(base + (size_t)n * ld + kk);
  return combine16(p[0], p[2]);
}

__device__ __forceinline__ v8f wmma_bf16(v16bf a, v16bf b, v8f c) {
  return __builtin_amdgcn_wmma_f32_16x16x32_bf16(false, a, false, b, (short)0, c, false, false);
}

// ---------------- fp32 -> bf16 weight conversion (once per launch) -------------
__global__ void cvtw_kernel(const float* __restrict__ src, bf16* __restrict__ dst, int n) {
  int i = blockIdx.x * 256 + threadIdx.x;
  if (i < n) dst[i] = (bf16)src[i];
}

// ---------------- LayerNorm rows of 128, fp32 in -> bf16 out -------------------
__global__ void ln_rows_kernel(const float* __restrict__ in, const float* __restrict__ g,
                               const float* __restrict__ b, bf16* __restrict__ out, int tin) {
  int wv = threadIdx.x >> 5, lane = threadIdx.x & 31;
  int r  = blockIdx.x * 8 + wv;
  int sr = tin ? ((r % NN) * NN + r / NN) : r;
  const float* row = in + (size_t)sr * CZ;
  float x[4]; float s = 0.f;
#pragma unroll
  for (int t = 0; t < 4; t++) { x[t] = row[lane + 32 * t]; s += x[t]; }
#pragma unroll
  for (int off = 16; off >= 1; off >>= 1) s += __shfl_xor(s, off, 32);
  float mu = s * (1.0f / CZ);
  float vr = 0.f;
#pragma unroll
  for (int t = 0; t < 4; t++) { float d = x[t] - mu; vr += d * d; }
#pragma unroll
  for (int off = 16; off >= 1; off >>= 1) vr += __shfl_xor(vr, off, 32);
  float inv = rsqrtf(vr * (1.0f / CZ) + 1e-5f);
#pragma unroll
  for (int t = 0; t < 4; t++) {
    int c = lane + 32 * t;
    out[(size_t)r * CZ + c] = (bf16)((x[t] - mu) * inv * g[c] + b[c]);
  }
}

// -------- gated dual projection: mask * sigmoid(zl@Wg^T+bg) * (zl@Wp^T+bp) -----
// writes bf16 scattered to [c][p][q] (tout=0, outgoing) or [c][q][p] (tout=1, incoming)
__global__ void proj_dual_kernel(const bf16* __restrict__ zl,
                                 const bf16* __restrict__ wg, const float* __restrict__ bgv,
                                 const bf16* __restrict__ wp, const float* __restrict__ bpv,
                                 const float* __restrict__ mask,
                                 bf16* __restrict__ outm, int tout) {
  int wv = threadIdx.x >> 5, lane = threadIdx.x & 31;
  int r0 = blockIdx.x * 16;
  int n0 = blockIdx.y * 64 + wv * 16;
  v8f ag = {0,0,0,0,0,0,0,0};
  v8f ap = {0,0,0,0,0,0,0,0};
#pragma unroll
  for (int k0 = 0; k0 < CZ; k0 += 32) {
    v16bf a = load_a_frag(zl, r0, CZ, k0);
    ag = wmma_bf16(a, load_b_frag(wg, n0, CZ, k0), ag);
    ap = wmma_bf16(a, load_b_frag(wp, n0, CZ, k0), ap);
  }
  int col = lane & 15, hi8 = (lane & 16) ? 8 : 0;
  int c = n0 + col;
  float big = bgv[c], bip = bpv[c];
#pragma unroll
  for (int v = 0; v < 8; v++) {
    int r = r0 + v + hi8;
    float val = mask[r] * sigf(ag[v] + big) * (ap[v] + bip);
    int pi = r / NN, qi = r - pi * NN;
    size_t idx = tout ? ((size_t)c * MM + (size_t)qi * NN + pi)
                      : ((size_t)c * MM + (size_t)pi * NN + qi);
    outm[idx] = (bf16)val;
  }
}

// -------- simple projection (16x32 tile/wave): mode 0 = plain*scale, 1 = sigmoid,
//          2 = write V^T layout -------------------------------------------------
__global__ void proj_simple_kernel(const bf16* __restrict__ x,
                                   const bf16* __restrict__ w, const float* __restrict__ bias,
                                   bf16* __restrict__ out, int mode, float scale) {
  int wv = threadIdx.x >> 5, lane = threadIdx.x & 31;
  int r0 = blockIdx.x * 16;
  int n0 = wv * 32;
  v8f acc0 = {0,0,0,0,0,0,0,0};
  v8f acc1 = {0,0,0,0,0,0,0,0};
#pragma unroll
  for (int k0 = 0; k0 < CZ; k0 += 32) {
    v16bf a = load_a_frag(x, r0, CZ, k0);
    acc0 = wmma_bf16(a, load_b_frag(w, n0,      CZ, k0), acc0);
    acc1 = wmma_bf16(a, load_b_frag(w, n0 + 16, CZ, k0), acc1);
  }
  int col = lane & 15, hi8 = (lane & 16) ? 8 : 0;
  int c0 = n0 + col, c1 = c0 + 16;
  float bv0 = bias ? bias[c0] : 0.0f;
  float bv1 = bias ? bias[c1] : 0.0f;
#pragma unroll
  for (int v = 0; v < 8; v++) {
    int r = r0 + v + hi8;
    float val0 = acc0[v] + bv0;
    float val1 = acc1[v] + bv1;
    if (mode == 1) { val0 = sigf(val0); val1 = sigf(val1); }
    val0 *= scale; val1 *= scale;
    if (mode == 2) {  // V^T layout: [i][h][c][key=j]
      int i = r / NN, j = r - i * NN;
      out[(((size_t)i * HH + (c0 >> 5)) * CA + (c0 & 31)) * NN + j] = (bf16)val0;
      out[(((size_t)i * HH + (c1 >> 5)) * CA + (c1 & 31)) * NN + j] = (bf16)val1;
    } else {
      out[(size_t)r * CZ + c0] = (bf16)val0;
      out[(size_t)r * CZ + c1] = (bf16)val1;
    }
  }
}

// -------- residual projection (16x32 tile/wave): z(+T) += [gate *] (x@W^T + b) --
__global__ void proj_resid_kernel(const bf16* __restrict__ x,
                                  const bf16* __restrict__ w, const float* __restrict__ bias,
                                  const bf16* __restrict__ gate,
                                  float* __restrict__ zb, int tout) {
  int wv = threadIdx.x >> 5, lane = threadIdx.x & 31;
  int r0 = blockIdx.x * 16;
  int n0 = wv * 32;
  v8f acc0 = {0,0,0,0,0,0,0,0};
  v8f acc1 = {0,0,0,0,0,0,0,0};
#pragma unroll
  for (int k0 = 0; k0 < CZ; k0 += 32) {
    v16bf a = load_a_frag(x, r0, CZ, k0);
    acc0 = wmma_bf16(a, load_b_frag(w, n0,      CZ, k0), acc0);
    acc1 = wmma_bf16(a, load_b_frag(w, n0 + 16, CZ, k0), acc1);
  }
  int col = lane & 15, hi8 = (lane & 16) ? 8 : 0;
  int c0 = n0 + col, c1 = c0 + 16;
  float bv0 = bias[c0], bv1 = bias[c1];
#pragma unroll
  for (int v = 0; v < 8; v++) {
    int r = r0 + v + hi8;
    float val0 = acc0[v] + bv0;
    float val1 = acc1[v] + bv1;
    if (gate) {
      val0 *= (float)gate[(size_t)r * CZ + c0];
      val1 *= (float)gate[(size_t)r * CZ + c1];
    }
    int pi = r / NN, qi = r - pi * NN;
    size_t base = tout ? ((size_t)qi * NN + pi) * CZ : (size_t)r * CZ;
    zb[base + c0] += val0;
    zb[base + c1] += val1;
  }
}

// -------- per-channel triangle einsum: X_c = A_c * B_c^T, 128 ch of 320x320x320 -
__global__ void einsum_kernel(const bf16* __restrict__ Am, const bf16* __restrict__ Bm,
                              float* __restrict__ x) {
  int c  = blockIdx.z;
  int it = blockIdx.x;
  int jt = blockIdx.y * 4 + (threadIdx.x >> 5);
  const bf16* A = Am + (size_t)c * MM;
  const bf16* B = Bm + (size_t)c * MM;
  v8f acc = {0,0,0,0,0,0,0,0};
#pragma unroll
  for (int k0 = 0; k0 < NN; k0 += 32) {
    if (k0 + 32 < NN) __builtin_prefetch(A + (size_t)(it * 16) * NN + k0 + 32, 0, 3);
    v16bf a = load_a_frag(A, it * 16, NN, k0);
    v16bf b = load_b_frag(B, jt * 16, NN, k0);
    acc = wmma_bf16(a, b, acc);
  }
  int lane = threadIdx.x & 31;
  int col = lane & 15, hi8 = (lane & 16) ? 8 : 0;
  int j = jt * 16 + col;
#pragma unroll
  for (int v = 0; v < 8; v++) {
    int i = it * 16 + v + hi8;
    x[((size_t)i * NN + j) * CZ + c] = acc[v];
  }
}

// -------- triangle bias: tb[h][p][q] = sum_c zl[p,q,c] * b_w[h,c] --------------
__global__ void tb_kernel(const bf16* __restrict__ zl, const float* __restrict__ bw,
                          float* __restrict__ tb) {
  int r = blockIdx.x * 64 + (threadIdx.x >> 2);
  int h = threadIdx.x & 3;
  const bf16* zr = zl + (size_t)r * CZ;
  const float* wr = bw + h * CZ;
  float acc = 0.f;
#pragma unroll 8
  for (int c = 0; c < CZ; c++) acc += (float)zr[c] * wr[c];
  tb[(size_t)h * MM + r] = acc;
}

// -------- flash attention per (i, h, 16-query tile), online softmax ------------
__global__ void flash_kernel(const bf16* __restrict__ qb, const bf16* __restrict__ kb,
                             const bf16* __restrict__ vT, const bf16* __restrict__ gb,
                             const float* __restrict__ tb, const float* __restrict__ mask,
                             bf16* __restrict__ ob, int mt) {
  __shared__ __attribute__((aligned(16))) bf16 stage[4][16 * 32];
  int wv = threadIdx.x >> 5, lane = threadIdx.x & 31;
  int i = blockIdx.x, h = blockIdx.y;
  int q0 = (blockIdx.z * 4 + wv) * 16;
  int col = lane & 15, hi8 = (lane & 16) ? 8 : 0;
  size_t rowb = (size_t)i * NN;

  v16bf qa = load_a_frag(qb, (int)rowb + q0, CZ, h * CA);   // Q tile, scale pre-folded
  v8f o0 = {0,0,0,0,0,0,0,0}, o1 = {0,0,0,0,0,0,0,0};
  float m[8], l[8];
#pragma unroll
  for (int v = 0; v < 8; v++) { m[v] = -1e30f; l[v] = 0.f; }
  bf16* st = stage[wv];

  for (int kt = 0; kt < NN; kt += 32) {
    v8f zf = {0,0,0,0,0,0,0,0};
    v8f s0 = wmma_bf16(qa, load_b_frag(kb, i * NN + kt,      CZ, h * CA), zf);
    v8f s1 = wmma_bf16(qa, load_b_frag(kb, i * NN + kt + 16, CZ, h * CA), zf);
    float mk0 = mt ? mask[(kt + col) * NN + i]      : mask[i * NN + kt + col];
    float mk1 = mt ? mask[(kt + 16 + col) * NN + i] : mask[i * NN + kt + 16 + col];
    float mb0 = 1e9f * (mk0 - 1.0f);
    float mb1 = 1e9f * (mk1 - 1.0f);
#pragma unroll
    for (int v = 0; v < 8; v++) {
      int qr = q0 + v + hi8;
      const float* tp = tb + (size_t)h * MM + (size_t)qr * NN + kt;
      s0[v] += mb0 + tp[col];
      s1[v] += mb1 + tp[16 + col];
      float rm = fmaxf(s0[v], s1[v]);
#pragma unroll
      for (int off = 8; off >= 1; off >>= 1) rm = fmaxf(rm, __shfl_xor(rm, off, 16));
      float mn = fmaxf(m[v], rm);
      float sc = __expf(m[v] - mn);
      m[v] = mn;
      float p0 = __expf(s0[v] - mn), p1 = __expf(s1[v] - mn);
      float rs = p0 + p1;
#pragma unroll
      for (int off = 8; off >= 1; off >>= 1) rs += __shfl_xor(rs, off, 16);
      l[v] = l[v] * sc + rs;
      o0[v] *= sc; o1[v] *= sc;
      // stash P in A-matrix memory order (16 rows x 32 keys) for re-load as A frag
      st[(v + hi8) * 32 + col]      = (bf16)p0;
      st[(v + hi8) * 32 + 16 + col] = (bf16)p1;
    }
    v16bf pa = load_a_frag(st, 0, 32, 0);
    int vrow = (i * HH + h) * CA;
    o0 = wmma_bf16(pa, load_b_frag(vT, vrow,      NN, kt), o0);
    o1 = wmma_bf16(pa, load_b_frag(vT, vrow + 16, NN, kt), o1);
  }
#pragma unroll
  for (int v = 0; v < 8; v++) {
    float inv = 1.0f / l[v];
    size_t row = rowb + q0 + v + hi8;
    size_t off0 = row * CZ + h * CA + col;
    ob[off0]      = (bf16)(o0[v] * inv * (float)gb[off0]);
    ob[off0 + 16] = (bf16)(o1[v] * inv * (float)gb[off0 + 16]);
  }
}

// ================================ host side ====================================
extern "C" void kernel_launch(void* const* d_in, const int* in_sizes, int n_in,
                              void* d_out, int out_size, void* d_ws, size_t ws_size,
                              hipStream_t stream) {
  (void)in_sizes; (void)n_in; (void)out_size; (void)ws_size;
  auto F = [&](int i) { return (const float*)d_in[i]; };

  const float* z_in = F(0);
  const float* mask = F(1);

  struct Tmu { const float *lig,*lib,*apw,*apb,*agw,*agb,*bpw,*bpb,*bgw,*bgb,*log,*lob,*zw,*zb,*gw,*gb; };
  auto getTmu = [&](int b) {
    Tmu t{F(b),F(b+1),F(b+2),F(b+3),F(b+4),F(b+5),F(b+6),F(b+7),
          F(b+8),F(b+9),F(b+10),F(b+11),F(b+12),F(b+13),F(b+14),F(b+15)};
    return t;
  };
  struct Att { const float *lg,*lb,*qw,*kw,*vw,*bw,*gw,*gb,*ow,*ob; };
  auto getAtt = [&](int b) {
    Att a{F(b),F(b+1),F(b+2),F(b+3),F(b+4),F(b+5),F(b+6),F(b+7),F(b+8),F(b+9)};
    return a;
  };
  Tmu t_out = getTmu(2);
  Tmu t_in  = getTmu(18);
  Att a_st  = getAtt(34);
  Att a_en  = getAtt(44);

  // workspace layout (bf16 M x 128 slot = 26.2 MB)
  const size_t SL = (size_t)MM * CZ * sizeof(bf16);
  char* ws = (char*)d_ws;
  bf16* s0   = (bf16*)(ws);            // zl / x_ln
  bf16* s1   = (bf16*)(ws + 1 * SL);   // A-mat / Q
  bf16* s2   = (bf16*)(ws + 2 * SL);   // B-mat / K
  bf16* s3   = (bf16*)(ws + 3 * SL);   // tri-mul gate / V^T
  float* xf  = (float*)(ws + 4 * SL);  // einsum output fp32 (2 slots)
  bf16* gatt = (bf16*)(ws + 4 * SL);   // attention gate (aliases xf region)
  bf16* obuf = (bf16*)(ws + 5 * SL);   // gated attention output
  float* tbp = (float*)(ws + 6 * SL);  // triangle bias [4][320][320] (1.64 MB)
  bf16* wpool = (bf16*)(ws + 6 * SL + (2u << 20));  // 22 x 32KB bf16 weights

  // pre-convert all 128x128 weight matrices to bf16 (weights stay L2-resident)
  int widx = 0;
  auto cw = [&](const float* src) {
    bf16* d = wpool + (size_t)widx * (CZ * CZ);
    widx++;
    cvtw_kernel<<<(CZ * CZ + 255) / 256, 256, 0, stream>>>(src, d, CZ * CZ);
    return (const bf16*)d;
  };
  struct TmuW { const bf16 *agw,*apw,*bgw,*bpw,*gw,*zw; };
  struct AttW { const bf16 *qw,*kw,*vw,*gw,*ow; };
  auto cvtTmu = [&](const Tmu& p) { TmuW w{cw(p.agw),cw(p.apw),cw(p.bgw),cw(p.bpw),cw(p.gw),cw(p.zw)}; return w; };
  auto cvtAtt = [&](const Att& p) { AttW w{cw(p.qw),cw(p.kw),cw(p.vw),cw(p.gw),cw(p.ow)}; return w; };
  TmuW w_out = cvtTmu(t_out);
  TmuW w_in  = cvtTmu(t_in);
  AttW w_st  = cvtAtt(a_st);
  AttW w_en  = cvtAtt(a_en);

  float* zbuf = (float*)d_out;
  hipMemcpyAsync(zbuf, z_in, (size_t)MM * CZ * sizeof(float),
                 hipMemcpyDeviceToDevice, stream);

  dim3 PG(MM / 16, 2, 1);      // dual-projection grid (wave = 16x16 tile x2 mats)
  dim3 PS(MM / 16, 1, 1);      // simple/resid grid (wave = 16x32 tile)
  dim3 EG(NN / 16, 5, CZ);     // einsum grid
  dim3 FG(NN, HH, 5);          // flash grid
  const float qscale = 0.1767766952966369f;  // 1/sqrt(32)

  auto tri_mul = [&](const Tmu& p, const TmuW& w, int incoming) {
    ln_rows_kernel<<<MM / 8, 256, 0, stream>>>(zbuf, p.lig, p.lib, s0, 0);
    proj_dual_kernel<<<PG, 128, 0, stream>>>(s0, w.agw, p.agb, w.apw, p.apb, mask, s1, incoming);
    proj_dual_kernel<<<PG, 128, 0, stream>>>(s0, w.bgw, p.bgb, w.bpw, p.bpb, mask, s2, incoming);
    proj_simple_kernel<<<PS, 128, 0, stream>>>(s0, w.gw, p.gb, s3, 1, 1.0f);
    einsum_kernel<<<EG, 128, 0, stream>>>(s1, s2, xf);
    ln_rows_kernel<<<MM / 8, 256, 0, stream>>>(xf, p.log, p.lob, s0, 0);
    proj_resid_kernel<<<PS, 128, 0, stream>>>(s0, w.zw, p.zb, s3, zbuf, 0);
  };

  auto tri_att = [&](const Att& p, const AttW& w, int transposed) {
    ln_rows_kernel<<<MM / 8, 256, 0, stream>>>(zbuf, p.lg, p.lb, s0, transposed);
    proj_simple_kernel<<<PS, 128, 0, stream>>>(s0, w.qw, nullptr, s1, 0, qscale);
    proj_simple_kernel<<<PS, 128, 0, stream>>>(s0, w.kw, nullptr, s2, 0, 1.0f);
    proj_simple_kernel<<<PS, 128, 0, stream>>>(s0, w.vw, nullptr, s3, 2, 1.0f);
    proj_simple_kernel<<<PS, 128, 0, stream>>>(s0, w.gw, p.gb, gatt, 1, 1.0f);
    tb_kernel<<<MM / 64, 256, 0, stream>>>(s0, p.bw, tbp);
    flash_kernel<<<FG, 128, 0, stream>>>(s1, s2, s3, gatt, tbp, mask, obuf, transposed);
    proj_resid_kernel<<<PS, 128, 0, stream>>>(obuf, w.ow, p.ob, nullptr, zbuf, transposed);
  };

  tri_mul(t_out, w_out, 0);   // outgoing
  tri_mul(t_in,  w_in,  1);   // incoming
  tri_att(a_st,  w_st,  0);   // starting node
  tri_att(a_en,  w_en,  1);   // ending node (operates on z^T via index flags)
}
